// SlidingWindowAttention_37873021616745
// MI455X (gfx1250) — matrix operand order repaired and checked by reading, not compiled, not run on hardware
//
#include <hip/hip_runtime.h>
#include <hip/hip_bf16.h>
#include <cstdint>

// ---------------------------------------------------------------------------
// SlidingWindowAttention on MI455X (gfx1250): bf16 WMMA + CDNA5 data movers.
//   x [4,4096,1024] f32, Wqkv [3072,1024] f32, Wproj [1024,1024] f32
//   out [4,4096,1024] f32
// Pipeline: cvt->bf16, QKV GEMM (async-to-LDS staging, scatter to
//           [bh][256][1024] q/k/v), per-(bh,slab) attention (TDM q-slab load
//           with HW LDS padding, S, softmax, P@V), proj GEMM -> f32 out.
// Workspace usage ~176 MB.
// ---------------------------------------------------------------------------

typedef __bf16 bf16;
typedef __attribute__((ext_vector_type(16))) __bf16 v16bf;
typedef __attribute__((ext_vector_type(8)))  __bf16 v8bf;
typedef __attribute__((ext_vector_type(4)))  __bf16 v4bf;
typedef __attribute__((ext_vector_type(8)))  float  v8f;
typedef __attribute__((ext_vector_type(4)))  unsigned int u32x4;
typedef __attribute__((ext_vector_type(8)))  int i32x8;
typedef __attribute__((ext_vector_type(4)))  int i32x4;

#define DIM_C     1024
#define N_HEADS   16
#define HEAD_D    64
#define WIN_      256
#define BATCH_    4
#define SEQ_      4096
#define ROWS_     (BATCH_*SEQ_)      // 16384 tokens
#define BH_       (BATCH_*N_HEADS)   // 64
#define DF_       1024               // flattened feature per window row
#define QKV_N     (3*DIM_C)          // 3072

// ---- CDNA5 sync/data-mover helpers ----------------------------------------
// Generic LDS pointers carry the LDS byte offset in their low 32 bits
static __device__ __forceinline__ unsigned lds_off(const void* p) {
  return (unsigned)(unsigned long long)(uintptr_t)p;
}

// GLOBAL_LOAD_ASYNC_TO_LDS_B128: per-lane 16B global -> LDS, ASYNCcnt-tracked.
static __device__ __forceinline__ void async_g2l_b128(unsigned lds_byte,
                                                      const void* gptr) {
  asm volatile("global_load_async_to_lds_b128 %0, %1, off"
               :: "v"(lds_byte), "v"((unsigned long long)(uintptr_t)gptr)
               : "memory");
}

static __device__ __forceinline__ void wait_async0() {
#if __has_builtin(__builtin_amdgcn_s_wait_asynccnt)
  __builtin_amdgcn_s_wait_asynccnt(0);
#else
  asm volatile("s_wait_asynccnt 0x0" ::: "memory");
#endif
}

static __device__ __forceinline__ void wait_tensor0() {
#if __has_builtin(__builtin_amdgcn_s_wait_tensorcnt)
  __builtin_amdgcn_s_wait_tensorcnt(0);
#else
  asm volatile("s_wait_tensorcnt 0x0" ::: "memory");
#endif
}

// Tensor Data Mover availability / arity (amdgpu-toolchain ships the TDM
// header and uses the 6-arg builtin; ROCm 7.2 has the 5-arg form).
#if __has_include(<hip/amd_detail/amd_gfx1250_TDM.h>)
#define TDM_ARITY 6
#elif __has_builtin(__builtin_amdgcn_tensor_load_to_lds)
#define TDM_ARITY 5
#else
#define TDM_ARITY 0
#endif

// q-slab TDM tile: 1024(f, contiguous) x 64(rows, stride 1024) bf16, with HW
// LDS padding of 4 DWORDs (8 bf16) after every 256 DWORDs (512 bf16).
// Resulting LDS row layout: elem (r,f) at r*1040 + f + (f>=512 ? 8 : 0).
#define QROW_ 1040

#if TDM_ARITY > 0
static __device__ __forceinline__ void tdm_load_q(unsigned lds_byte,
                                                  const bf16* gaddr) {
  unsigned long long ga = (unsigned long long)(uintptr_t)gaddr;
  u32x4 g0;
  g0[0] = 1u;                                       // count=1, user descriptor
  g0[1] = lds_byte;                                 // lds_addr [63:32]
  g0[2] = (unsigned)(ga & 0xffffffffull);           // global_addr lo
  g0[3] = (unsigned)((ga >> 32) & 0x1ffffffull)     // global_addr [56:32]
          | (2u << 30);                             // type=2 ("image")
  // group1 bitfields (see ISA 8.4), packed via u64 then split
  unsigned long long u0 = (1ull << 16)              // data_size = 2B
                        | (1ull << 20)              // pad_enable
                        | (7ull << 22)              // pad_interval: 256 DWORDs
                        | (3ull << 25)              // pad_amount: 4 DWORDs
                        | (1024ull << 48);          // tensor_dim0 lo16
  unsigned long long u1 = (1024ull >> 16)           // tensor_dim0 hi16
                        | (64ull << 16)             // tensor_dim1
                        | (1024ull << 48);          // tile_dim0
  unsigned long long u2 = 64ull                     // tile_dim1
                        | (0ull << 16)              // tile_dim2
                        | ((1024ull & 0xffffffffull) << 32); // dim0_stride lo
  unsigned long long u3 = (1024ull >> 32)           // dim0_stride hi
                        | (0ull << 16);             // dim1_stride (unused 2D)
  i32x8 g1;
  g1[0] = (int)u0; g1[1] = (int)(u0 >> 32);
  g1[2] = (int)u1; g1[3] = (int)(u1 >> 32);
  g1[4] = (int)u2; g1[5] = (int)(u2 >> 32);
  g1[6] = (int)u3; g1[7] = (int)(u3 >> 32);
  i32x4 z4 = {0, 0, 0, 0};
#if TDM_ARITY == 6
  i32x8 z8 = {0, 0, 0, 0, 0, 0, 0, 0};
  __builtin_amdgcn_tensor_load_to_lds(g0, g1, z4, z4, z8, 0);
#else
  __builtin_amdgcn_tensor_load_to_lds(g0, g1, z4, z4, 0);
#endif
}
#endif

// ---- WMMA wrapper ---------------------------------------------------------
static __device__ __forceinline__ v8f wmma_bf16f32(v16bf a, v16bf b, v8f c) {
  return __builtin_amdgcn_wmma_f32_16x16x32_bf16(false, a, false, b,
                                                 (short)0, c, false, false);
}

// A fragment (16x32 bf16) from a row-major buffer, row stride `str` elems.
static __device__ __forceinline__ v16bf frag_a_bf16(const bf16* __restrict__ base,
                                                    int str, int m0, int k0, int lane) {
  int row = m0 + (lane & 15);
  int klo = (lane >> 4) << 3;
  const bf16* p = base + (size_t)row * str + k0 + klo;
  union { v16bf v; v8bf h[2]; } u;
  u.h[0] = *(const v8bf*)p;          // K = k0+klo   .. +7
  u.h[1] = *(const v8bf*)(p + 16);   // K = k0+16+klo.. +7
  return u.v;
}

// A fragment from the TDM-padded q slab (row stride QROW_, +8 gap at f=512).
static __device__ __forceinline__ v16bf frag_a_qpad(const bf16* __restrict__ base,
                                                    int m0, int k0, int lane) {
  int row = m0 + (lane & 15);
  int klo = (lane >> 4) << 3;
  int f0 = k0 + klo, f1 = f0 + 16;   // 8-elem chunks never straddle f=512
  const bf16* p0 = base + (size_t)row * QROW_ + f0 + ((f0 >= 512) ? 8 : 0);
  const bf16* p1 = base + (size_t)row * QROW_ + f1 + ((f1 >= 512) ? 8 : 0);
  union { v16bf v; v8bf h[2]; } u;
  u.h[0] = *(const v8bf*)p0;
  u.h[1] = *(const v8bf*)p1;
  return u.v;
}

// B fragment (32x16): B[k][n] = src[n][k], src row-major [N][K] stride `str`.
static __device__ __forceinline__ v16bf frag_b_bf16(const bf16* __restrict__ base,
                                                    int str, int n0, int k0, int lane) {
  int n  = n0 + (lane & 15);
  int kh = (lane >> 4) << 3;
  const bf16* p = base + (size_t)n * str + k0 + kh;
  union { v16bf v; v8bf h[2]; } u;
  u.h[0] = *(const v8bf*)p;
  u.h[1] = *(const v8bf*)(p + 16);
  return u.v;
}

// A fragment built from f32 LDS (softmax probabilities), converted to bf16.
static __device__ __forceinline__ v16bf frag_a_f32(const float* __restrict__ base,
                                                   int str, int m0, int k0, int lane) {
  int row = m0 + (lane & 15);
  int klo = (lane >> 4) << 3;
  const float* p = base + (size_t)row * str + k0 + klo;
  v16bf r = {};
#pragma unroll
  for (int i = 0; i < 8; ++i) {
    r[i]     = (__bf16)p[i];
    r[i + 8] = (__bf16)p[i + 16];
  }
  return r;
}

// ---- f32 -> bf16 conversion (vectorized) ----------------------------------
__global__ void __launch_bounds__(256) swa_cvt_bf16(const float* __restrict__ src,
                                                    bf16* __restrict__ dst, int n4) {
  int i = blockIdx.x * blockDim.x + threadIdx.x;
  if (i < n4) {
    float4 f = ((const float4*)src)[i];
    v4bf o;
    o[0] = (__bf16)f.x; o[1] = (__bf16)f.y; o[2] = (__bf16)f.z; o[3] = (__bf16)f.w;
    ((v4bf*)dst)[i] = o;
  }
}

// ---- generic bf16 GEMM: C = A[M,K] @ W[N,K]^T -----------------------------
// 128x128 block tile, BK=32, async global->LDS staging (ASYNCcnt).
// MODE 0: QKV projection -> scatter bf16 into q/k/v [bh][256][1024], q scaled.
// MODE 1: output projection -> f32 row-major out.
#define BKK  32
#define LSTR 40   // 32 + 8 pad (80 B rows: 16B-aligned, bank-friendly)

template <int MODE>
__global__ void __launch_bounds__(256) swa_gemm_bf16(
    const bf16* __restrict__ A, const bf16* __restrict__ W,
    int K, int lda, int ldw, int Ntot,
    bf16* __restrict__ qb, bf16* __restrict__ kbuf, bf16* __restrict__ vbuf,
    float* __restrict__ outp)
{
  __shared__ bf16 sA[128 * LSTR];
  __shared__ bf16 sB[128 * LSTR];

  const int tid = threadIdx.x, lane = tid & 31, wv = tid >> 5;
  const int bm = blockIdx.y << 7, bn = blockIdx.x << 7;
  const int m0 = (wv >> 1) << 5;     // 0,32,64,96
  const int n0 = (wv & 1) << 6;      // 0,64

  v8f acc[8] = {};

  const int srow = tid >> 1, shalf = (tid & 1) << 4;
  const bf16* ga = A + (size_t)(bm + srow) * lda + shalf;
  const bf16* gw = W + (size_t)(bn + srow) * ldw + shalf;
  const unsigned ldsA = lds_off(&sA[srow * LSTR + shalf]);
  const unsigned ldsB = lds_off(&sB[srow * LSTR + shalf]);

  for (int kk = 0; kk < K; kk += BKK) {
    if (kk + BKK < K) {                       // global_prefetch_b8 next panels
      __builtin_prefetch(ga + kk + BKK, 0, 0);
      __builtin_prefetch(gw + kk + BKK, 0, 0);
    }
    __syncthreads();                          // prior-iter LDS readers done
    async_g2l_b128(ldsA,      ga + kk);       // 16B x2 per array, per thread
    async_g2l_b128(ldsA + 16, ga + kk + 8);
    async_g2l_b128(ldsB,      gw + kk);
    async_g2l_b128(ldsB + 16, gw + kk + 8);
    wait_async0();                            // own copies complete
    __syncthreads();                          // all waves' copies visible

    v16bf af[2], bfv[4];
#pragma unroll
    for (int mi = 0; mi < 2; ++mi) af[mi] = frag_a_bf16(sA, LSTR, m0 + mi * 16, 0, lane);
#pragma unroll
    for (int ni = 0; ni < 4; ++ni) bfv[ni] = frag_b_bf16(sB, LSTR, n0 + ni * 16, 0, lane);
#pragma unroll
    for (int mi = 0; mi < 2; ++mi)
#pragma unroll
      for (int ni = 0; ni < 4; ++ni)
        acc[mi * 4 + ni] = wmma_bf16f32(af[mi], bfv[ni], acc[mi * 4 + ni]);
  }

  // Epilogue. C layout: elem r -> row = m0+r+8*(lane>=16), col = n0+lane%16.
  const int hi = (lane >> 4) << 3, nc = lane & 15;
#pragma unroll
  for (int mi = 0; mi < 2; ++mi)
#pragma unroll
    for (int ni = 0; ni < 4; ++ni) {
      v8f c = acc[mi * 4 + ni];
#pragma unroll
      for (int r = 0; r < 8; ++r) {
        int m = bm + m0 + mi * 16 + hi + r;
        int n = bn + n0 + ni * 16 + nc;
        float val = c[r];
        if constexpr (MODE == 0) {
          int three = n >> 10, rem = n & 1023;
          int h = rem >> 6, d = rem & 63;
          int b = m >> 12, t = m & 4095;
          int w = t >> 4, f = ((t & 15) << 6) | d;   // window row / flat feature
          bf16* dst = (three == 0) ? qb : (three == 1 ? kbuf : vbuf);
          if (three == 0) val *= 0.125f;             // 1/sqrt(head_dim=64)
          dst[((size_t)(b * N_HEADS + h) * WIN_ + w) * DF_ + f] = (bf16)val;
        } else {
          outp[(size_t)m * Ntot + n] = val;
        }
      }
    }
}

// ---- attention: one WG per (bh, 64-row slab of the 256x256 S matrix) ------
#define SSTR 264                     // 256 + 8 pad (f32)
#define VSTR 272                     // 256 + 16 pad (bf16)
#define REGA_BYTES 139264            // max(64*QROW_*2 = 133120, 256*VSTR*2)
#define ATTN_SMEM (REGA_BYTES + 64 * SSTR * 4)   // 206848 B (< 320 KB/WG)

__global__ void __launch_bounds__(256) swa_attn_kernel(
    const bf16* __restrict__ qb, const bf16* __restrict__ kbuf,
    const bf16* __restrict__ vbuf, bf16* __restrict__ aob)
{
  extern __shared__ char smem[];
  bf16*  sQ  = (bf16*)smem;                 // phase 1: 64x1024 q slab (padded)
  bf16*  sVt = (bf16*)smem;                 // phase 2: transposed V panel
  float* sS  = (float*)(smem + REGA_BYTES); // 64x256 scores/probs (f32)

  const int tid = threadIdx.x, lane = tid & 31, wv = tid >> 5;
  const int bh = blockIdx.y, slab = blockIdx.x;
  const bf16* qg = qb   + ((size_t)bh * WIN_ + slab * 64) * DF_;
  const bf16* kg = kbuf + (size_t)bh * WIN_ * DF_;
  const bf16* vg = vbuf + (size_t)bh * WIN_ * DF_;

  // stage q slab 64x1024 via Tensor Data Mover (wave 0 issues the DMA;
  // HW pads LDS rows to QROW_=1040 elems with a gap at f=512).
#if TDM_ARITY > 0
  if (tid < 32) {
    tdm_load_q(lds_off(sQ), qg);
    wait_tensor0();
  }
#else
  for (int idx = tid; idx < 64 * (DF_ / 8); idx += 256) {
    int row = idx >> 7, c = (idx & 127) << 3;
    *(v8bf*)&sQ[(size_t)row * QROW_ + c + ((c >= 512) ? 8 : 0)] =
        *(const v8bf*)&qg[(size_t)row * DF_ + c];
  }
#endif
  __syncthreads();

  // ---- S = q @ k^T (scale pre-folded into q). Wave: 2 n-tiles x 4 m-tiles.
  {
    v8f acc[8] = {};
    const int nt0 = wv << 1;
    for (int kk = 0; kk < DF_; kk += 32) {
      v16bf af[4], bfv[2];
#pragma unroll
      for (int mi = 0; mi < 4; ++mi) af[mi] = frag_a_qpad(sQ, mi * 16, kk, lane);
#pragma unroll
      for (int j = 0; j < 2; ++j) bfv[j] = frag_b_bf16(kg, DF_, (nt0 + j) * 16, kk, lane);
#pragma unroll
      for (int mi = 0; mi < 4; ++mi)
#pragma unroll
        for (int j = 0; j < 2; ++j)
          acc[mi * 2 + j] = wmma_bf16f32(af[mi], bfv[j], acc[mi * 2 + j]);
    }
    const int hi = (lane >> 4) << 3, nc = lane & 15;
#pragma unroll
    for (int mi = 0; mi < 4; ++mi)
#pragma unroll
      for (int j = 0; j < 2; ++j)
#pragma unroll
        for (int r = 0; r < 8; ++r)
          sS[(mi * 16 + hi + r) * SSTR + (nt0 + j) * 16 + nc] = acc[mi * 2 + j][r];
  }
  __syncthreads();

  // ---- softmax over each of 64 rows (256 cols), one wave per 8 rows
  for (int r = wv * 8; r < wv * 8 + 8; ++r) {
    float* row = sS + (size_t)r * SSTR;
    float v[8], mx = -3.0e38f;
#pragma unroll
    for (int j = 0; j < 8; ++j) { v[j] = row[lane + (j << 5)]; mx = fmaxf(mx, v[j]); }
    for (int o = 16; o > 0; o >>= 1) mx = fmaxf(mx, __shfl_xor(mx, o, 32));
    float s = 0.f;
#pragma unroll
    for (int j = 0; j < 8; ++j) { v[j] = __expf(v[j] - mx); s += v[j]; }
    for (int o = 16; o > 0; o >>= 1) s += __shfl_xor(s, o, 32);
    float inv = 1.0f / s;
#pragma unroll
    for (int j = 0; j < 8; ++j) row[lane + (j << 5)] = v[j] * inv;
  }
  __syncthreads();

  // ---- O = P @ V, 4 passes over 256-wide feature panels
  const int b = bh >> 4, h = bh & 15;
  for (int p = 0; p < 4; ++p) {
    const int f0 = p << 8;
    {   // stage transposed V panel: sVt[f - f0][x], one V row per thread
      const bf16* vrow = vg + (size_t)tid * DF_ + f0;
      for (int c = 0; c < 256; c += 8) {
        v8bf vv = *(const v8bf*)(vrow + c);
#pragma unroll
        for (int e = 0; e < 8; ++e) sVt[(c + e) * VSTR + tid] = vv[e];
      }
    }
    __syncthreads();

    v8f oacc[8] = {};
    const int nt0 = wv << 1;     // 2 of this pass's 16 f-tiles per wave
    for (int xk = 0; xk < WIN_; xk += 32) {
      v16bf af[4], bfv[2];
#pragma unroll
      for (int mi = 0; mi < 4; ++mi) af[mi] = frag_a_f32(sS, SSTR, mi * 16, xk, lane);
#pragma unroll
      for (int j = 0; j < 2; ++j) bfv[j] = frag_b_bf16(sVt, VSTR, (nt0 + j) * 16, xk, lane);
#pragma unroll
      for (int mi = 0; mi < 4; ++mi)
#pragma unroll
        for (int j = 0; j < 2; ++j)
          oacc[mi * 2 + j] = wmma_bf16f32(af[mi], bfv[j], oacc[mi * 2 + j]);
    }

    // scatter to token-major attention output [B*N, 1024] (bf16)
    const int hi = (lane >> 4) << 3, nc = lane & 15;
#pragma unroll
    for (int mi = 0; mi < 4; ++mi)
#pragma unroll
      for (int j = 0; j < 2; ++j)
#pragma unroll
        for (int r = 0; r < 8; ++r) {
          int w = (slab << 6) + mi * 16 + hi + r;   // window row 0..255
          int f = f0 + (nt0 + j) * 16 + nc;         // flat feature 0..1023
          int t = (w << 4) | (f >> 6);
          int d = f & 63;
          aob[((size_t)(b * SEQ_ + t)) * DIM_C + (h << 6) + d] =
              (bf16)oacc[mi * 2 + j][r];
        }
    __syncthreads();   // sVt consumed before next pass restages
  }
}

// ---------------------------------------------------------------------------
extern "C" void kernel_launch(void* const* d_in, const int* in_sizes, int n_in,
                              void* d_out, int out_size, void* d_ws, size_t ws_size,
                              hipStream_t stream) {
  (void)in_sizes; (void)n_in; (void)out_size; (void)ws_size; // needs ~176 MB ws
  const float* x    = (const float*)d_in[0];
  const float* Wqkv = (const float*)d_in[1];
  const float* Wprj = (const float*)d_in[2];
  float* out = (float*)d_out;

  char* ws = (char*)d_ws;
  size_t off = 0;
  auto carve = [&](size_t bytes) -> char* {
    char* p = ws + off;
    off += (bytes + 255) & ~(size_t)255;
    return p;
  };
  bf16* xb   = (bf16*)carve((size_t)ROWS_ * DIM_C * 2);   // 32 MB
  bf16* wqb  = (bf16*)carve((size_t)QKV_N * DIM_C * 2);   // 6 MB
  bf16* wpb  = (bf16*)carve((size_t)DIM_C * DIM_C * 2);   // 2 MB
  bf16* qb   = (bf16*)carve((size_t)BH_ * WIN_ * DF_ * 2);// 32 MB
  bf16* kbuf = (bf16*)carve((size_t)BH_ * WIN_ * DF_ * 2);// 32 MB
  bf16* vbuf = (bf16*)carve((size_t)BH_ * WIN_ * DF_ * 2);// 32 MB
  bf16* aob  = (bf16*)carve((size_t)ROWS_ * DIM_C * 2);   // 32 MB

  // 1) f32 -> bf16 conversions
  int n4;
  n4 = ROWS_ * DIM_C / 4;
  swa_cvt_bf16<<<(n4 + 255) / 256, 256, 0, stream>>>(x, xb, n4);
  n4 = QKV_N * DIM_C / 4;
  swa_cvt_bf16<<<(n4 + 255) / 256, 256, 0, stream>>>(Wqkv, wqb, n4);
  n4 = DIM_C * DIM_C / 4;
  swa_cvt_bf16<<<(n4 + 255) / 256, 256, 0, stream>>>(Wprj, wpb, n4);

  // 2) QKV projection, scattered straight into attention layout
  swa_gemm_bf16<0><<<dim3(QKV_N / 128, ROWS_ / 128), 256, 0, stream>>>(
      xb, wqb, DIM_C, DIM_C, DIM_C, QKV_N, qb, kbuf, vbuf, nullptr);

  // 3) attention (dynamic LDS ~202 KB; CDNA5 allows up to 320 KB/WG)
  hipFuncSetAttribute((const void*)swa_attn_kernel,
                      hipFuncAttributeMaxDynamicSharedMemorySize, ATTN_SMEM);
  swa_attn_kernel<<<dim3(WIN_ / 64, BH_), 256, ATTN_SMEM, stream>>>(qb, kbuf, vbuf, aob);

  // 4) output projection -> f32
  swa_gemm_bf16<1><<<dim3(DIM_C / 128, ROWS_ / 128), 256, 0, stream>>>(
      aob, wpb, DIM_C, DIM_C, DIM_C, DIM_C, nullptr, nullptr, nullptr, out);
}
